// Model_45406394253760
// MI455X (gfx1250) — matrix operand order repaired and checked by reading, not compiled, not run on hardware
//
#include <hip/hip_runtime.h>
#include <math.h>

// ---------------------------------------------------------------------------
// Problem constants
// ---------------------------------------------------------------------------
#define HDIM   1024
#define EMDIM  512
#define G4     4096          // 4*H
#define TLEN   2048
#define OSIZE  32000
#define START_IDX 1
#define LSTM_BLOCKS 64

typedef __attribute__((ext_vector_type(16))) __bf16 bf16x16;
typedef __attribute__((ext_vector_type(8)))  __bf16 bf16x8;
typedef __attribute__((ext_vector_type(8)))  float  f32x8;

// ---------------------------------------------------------------------------
// CDNA5 async global->LDS copy (ASYNCcnt-tracked, no VGPR staging).
// LDS operand: wave-relative LDS byte address = low 32 bits of generic ptr
// (ISA 10.2, LDS aperture).
// ---------------------------------------------------------------------------
__device__ __forceinline__ void async_ld16(void* lds_ptr, const void* gptr) {
    unsigned lds = (unsigned)(uintptr_t)lds_ptr;
    unsigned long long ga = (unsigned long long)(uintptr_t)gptr;
    asm volatile("global_load_async_to_lds_b128 %0, %1, off"
                 :: "v"(lds), "v"(ga) : "memory");
}
__device__ __forceinline__ void wait_async_le5() {
    asm volatile("s_wait_asynccnt 0x5" ::: "memory");
}
__device__ __forceinline__ void wait_async_0() {
    asm volatile("s_wait_asynccnt 0x0" ::: "memory");
}

// ---------------------------------------------------------------------------
// Elementwise fp32 -> bf16 conversion (weights)
// ---------------------------------------------------------------------------
__global__ __launch_bounds__(256) void f2bf_kernel(const float* __restrict__ in,
                                                   __bf16* __restrict__ out, long n) {
    long i = (long)blockIdx.x * blockDim.x + threadIdx.x;
    long stride = (long)gridDim.x * blockDim.x;
    for (; i < n; i += stride) out[i] = (__bf16)in[i];
}

// ---------------------------------------------------------------------------
// Embedding gathers (fp32 table -> bf16 rows)
// ---------------------------------------------------------------------------
__global__ __launch_bounds__(256) void gather_enc_kernel(const float* __restrict__ emb,
                                                         const int* __restrict__ ids,
                                                         __bf16* __restrict__ out) {
    int t = blockIdx.x;
    int id = ids[t];
    for (int e = threadIdx.x; e < EMDIM; e += blockDim.x)
        out[(size_t)t * EMDIM + e] = (__bf16)emb[(size_t)id * EMDIM + e];
}

__global__ __launch_bounds__(256) void gather_dec_kernel(const float* __restrict__ emb,
                                                         const int* __restrict__ gt,
                                                         __bf16* __restrict__ out) {
    int t = blockIdx.x;
    int id = (t == 0) ? START_IDX : gt[t - 1];
    for (int e = threadIdx.x; e < EMDIM; e += blockDim.x)
        out[(size_t)t * EMDIM + e] = (__bf16)emb[(size_t)id * EMDIM + e];
}

// ---------------------------------------------------------------------------
// State / barrier init, decoder state copy
// ---------------------------------------------------------------------------
__global__ __launch_bounds__(256) void init_state_kernel(int* bar, float* h2, float* c) {
    int i = blockIdx.x * blockDim.x + threadIdx.x;
    if (i < 2) bar[i] = 0;
    if (i < 2 * HDIM) h2[i] = 0.f;
    if (i < HDIM) c[i] = 0.f;
}

__global__ __launch_bounds__(256) void dec_init_kernel(const float* __restrict__ c,
                                                       float* __restrict__ h2) {
    int i = blockIdx.x * blockDim.x + threadIdx.x;
    if (i < HDIM) { h2[i] = c[i]; h2[HDIM + i] = c[i]; }
}

// ---------------------------------------------------------------------------
// WMMA GEMM:  C[M,N] = A[M,K](bf16) * B[N,K](bf16)^T + bias0[N] + bias1[N]
// Block = 256 threads (8 waves). Block tile 64x256, wave tile 32x64
// (2x4 WMMA 16x16 tiles -> 8 v_wmma per K-step). Double-buffered LDS tiles
// filled with async global->LDS DMA, overlapped with WMMA compute.
// Fragment layouts per CDNA5 ISA 7.12.2 (wave32).
// ---------------------------------------------------------------------------
__global__ __launch_bounds__(256) void wmma_gemm_bf16(const __bf16* __restrict__ A,
                                                      const __bf16* __restrict__ B,
                                                      const float* __restrict__ bias0,
                                                      const float* __restrict__ bias1,
                                                      float* __restrict__ C,
                                                      int M, int N, int K) {
    constexpr int BM = 64, BN = 256, BK = 32, PAD = 8, LDT = BK + PAD; // 80B row stride
    __shared__ __bf16 As[2][BM * LDT];
    __shared__ __bf16 Bs[2][BN * LDT];

    const int tid  = threadIdx.x;
    const int m0   = blockIdx.y * BM;
    const int n0   = blockIdx.x * BN;
    const int wave = tid >> 5, lane = tid & 31;
    const int wm   = (wave >> 2) * 32;       // 0 / 32
    const int wn   = (wave & 3) * 64;        // 0 / 64 / 128 / 192
    const int half = lane >> 4, r = lane & 15;

    // per-thread staging assignment: A = 1 x 16B chunk, B = 4 x 16B chunks
    const int arow = tid >> 2, acol = (tid & 3) * 8;
    const __bf16* agp = A + (size_t)(m0 + arow) * K + acol;
    const __bf16* bgp[4];
    int brow[4], bcol[4];
    #pragma unroll
    for (int i = 0; i < 4; ++i) {
        int cid = tid + i * 256;
        brow[i] = cid >> 2; bcol[i] = (cid & 3) * 8;
        bgp[i] = B + (size_t)(n0 + brow[i]) * K + bcol[i];
    }

    auto issue_tile = [&](int kt, int buf) {
        int ko = kt * BK;
        async_ld16(&As[buf][arow * LDT + acol], agp + ko);
        #pragma unroll
        for (int i = 0; i < 4; ++i)
            async_ld16(&Bs[buf][brow[i] * LDT + bcol[i]], bgp[i] + ko);
    };

    f32x8 acc[2][4] = {};
    const int KT = K / BK;

    issue_tile(0, 0);
    for (int kt = 0; kt < KT; ++kt) {
        const int buf = kt & 1;
        if (kt + 1 < KT) { issue_tile(kt + 1, buf ^ 1); wait_async_le5(); }
        else             { wait_async_0(); }
        __syncthreads();

        const __bf16* as = As[buf];
        const __bf16* bs = Bs[buf];

        // A fragments: lane half h -> K {h*8..h*8+7, 16+h*8..16+h*8+7}
        bf16x16 afr[2], bfr[4];
        #pragma unroll
        for (int mi = 0; mi < 2; ++mi) {
            const __bf16* p = as + (wm + mi * 16 + r) * LDT;
            bf16x8 lo = *(const bf16x8*)(p + half * 8);
            bf16x8 hi = *(const bf16x8*)(p + 16 + half * 8);
            afr[mi] = __builtin_shufflevector(lo, hi, 0,1,2,3,4,5,6,7,8,9,10,11,12,13,14,15);
        }
        // B fragments: lane half h -> K {h*8..h*8+7, 16+h*8..16+h*8+7} for column r
        #pragma unroll
        for (int ni = 0; ni < 4; ++ni) {
            const __bf16* p = bs + (wn + ni * 16 + r) * LDT;
            bf16x8 lo = *(const bf16x8*)(p + half * 8);
            bf16x8 hi = *(const bf16x8*)(p + 16 + half * 8);
            bfr[ni] = __builtin_shufflevector(lo, hi, 0,1,2,3,4,5,6,7,8,9,10,11,12,13,14,15);
        }
        #pragma unroll
        for (int mi = 0; mi < 2; ++mi)
            #pragma unroll
            for (int ni = 0; ni < 4; ++ni)
                acc[mi][ni] = __builtin_amdgcn_wmma_f32_16x16x32_bf16(
                    false, afr[mi], false, bfr[ni], (short)0, acc[mi][ni], false, false);
        __syncthreads();
    }

    // write-back: VGPR v -> M = v + half*8, N = r  (per ISA C/D layout)
    #pragma unroll
    for (int mi = 0; mi < 2; ++mi) {
        #pragma unroll
        for (int ni = 0; ni < 4; ++ni) {
            int col = n0 + wn + ni * 16 + r;
            float bv = (bias0 ? bias0[col] : 0.f) + (bias1 ? bias1[col] : 0.f);
            #pragma unroll
            for (int v = 0; v < 8; ++v) {
                int rowg = m0 + wm + mi * 16 + half * 8 + v;
                C[(size_t)rowg * N + col] = acc[mi][ni][v] + bv;
            }
        }
    }
}

// ---------------------------------------------------------------------------
// Device-wide barrier (generation counter, agent scope, s_sleep backoff)
// ---------------------------------------------------------------------------
__device__ __forceinline__ void grid_barrier(int* bar, int nblocks) {
    __syncthreads();
    if (threadIdx.x == 0) {
        __threadfence();
        int g = __hip_atomic_load(&bar[1], __ATOMIC_ACQUIRE, __HIP_MEMORY_SCOPE_AGENT);
        if (__hip_atomic_fetch_add(&bar[0], 1, __ATOMIC_ACQ_REL, __HIP_MEMORY_SCOPE_AGENT)
            == nblocks - 1) {
            __hip_atomic_store(&bar[0], 0, __ATOMIC_RELAXED, __HIP_MEMORY_SCOPE_AGENT);
            __hip_atomic_fetch_add(&bar[1], 1, __ATOMIC_ACQ_REL, __HIP_MEMORY_SCOPE_AGENT);
        } else {
            while (__hip_atomic_load(&bar[1], __ATOMIC_ACQUIRE, __HIP_MEMORY_SCOPE_AGENT) == g)
                __builtin_amdgcn_s_sleep(2);
        }
    }
    __syncthreads();
}

__device__ __forceinline__ float sigmoidf_(float x) { return 1.f / (1.f + expf(-x)); }

// ---------------------------------------------------------------------------
// Persistent LSTM: 64 blocks x 256 threads, block b owns hidden units
// [b*16, b*16+16) -> 64 gate rows of Whh. h ping-pongs between two buffers;
// one grid barrier per timestep. x_gates precomputed by WMMA GEMM.
// ---------------------------------------------------------------------------
__global__ __launch_bounds__(256) void lstm_persistent_kernel(
    const float* __restrict__ xg,     // [T, 4H]
    const __bf16* __restrict__ Whh,   // [4H, H] bf16
    float* __restrict__ h2,           // [2, H] ping-pong hidden
    float* __restrict__ c,            // [H]
    __bf16* __restrict__ hs_out,      // [T, H] or nullptr
    int T, int* bar) {
    __shared__ float hsh[HDIM];
    __shared__ float gbuf[64];

    const int tid = threadIdx.x, b = blockIdx.x;
    const int wave = tid >> 5, lane = tid & 31;

    for (int t = 0; t < T; ++t) {
        const float* hread = h2 + (t & 1) * HDIM;
        float* hwrite      = h2 + ((t + 1) & 1) * HDIM;

        for (int i = tid; i < HDIM; i += 256) hsh[i] = hread[i];
        __syncthreads();

        // each wave: 8 gate rows, bf16 weights * fp32 h (LDS), wave32 reduce
        for (int rr = 0; rr < 8; ++rr) {
            int idx = wave * 8 + rr;                 // 0..63
            int g = idx >> 4;                        // gate: i,f,g,o
            int u = (b << 4) + (idx & 15);           // hidden unit
            int row = g * HDIM + u;
            const __bf16* w = Whh + (size_t)row * HDIM;
            float sum = 0.f;
            #pragma unroll
            for (int it = 0; it < 4; ++it) {
                int kb = it * 256 + lane * 8;
                bf16x8 wv = *(const bf16x8*)(w + kb);
                #pragma unroll
                for (int j = 0; j < 8; ++j) sum += (float)wv[j] * hsh[kb + j];
            }
            #pragma unroll
            for (int off = 16; off > 0; off >>= 1) sum += __shfl_xor(sum, off, 32);
            if (lane == 0) gbuf[idx] = sum + xg[(size_t)t * G4 + row];
        }
        __syncthreads();

        if (tid < 16) {
            int u = (b << 4) + tid;
            float ig = gbuf[tid], fg = gbuf[16 + tid];
            float gg = gbuf[32 + tid], og = gbuf[48 + tid];
            float cn = sigmoidf_(fg) * c[u] + sigmoidf_(ig) * tanhf(gg);
            float hn = sigmoidf_(og) * tanhf(cn);
            c[u] = cn;
            hwrite[u] = hn;
            if (hs_out) hs_out[(size_t)t * HDIM + u] = (__bf16)hn;
        }
        grid_barrier(bar, gridDim.x);
    }
}

// ---------------------------------------------------------------------------
// Host driver
// ---------------------------------------------------------------------------
extern "C" void kernel_launch(void* const* d_in, const int* in_sizes, int n_in,
                              void* d_out, int out_size, void* d_ws, size_t ws_size,
                              hipStream_t stream) {
    const int*   e1      = (const int*)d_in[0];
    const int*   gt      = (const int*)d_in[1];
    const float* emb_i   = (const float*)d_in[2];
    const float* emb_o   = (const float*)d_in[3];
    const float* enc_Wih = (const float*)d_in[4];
    const float* enc_Whh = (const float*)d_in[5];
    const float* enc_bih = (const float*)d_in[6];
    const float* enc_bhh = (const float*)d_in[7];
    const float* dec_Wih = (const float*)d_in[8];
    const float* dec_Whh = (const float*)d_in[9];
    const float* dec_bih = (const float*)d_in[10];
    const float* dec_bhh = (const float*)d_in[11];
    const float* outW    = (const float*)d_in[12];
    const float* outb    = (const float*)d_in[13];
    float* out = (float*)d_out;

    char* ws = (char*)d_ws;
    size_t off = 0;
    auto take = [&](size_t bytes) -> char* {
        char* p = ws + off;
        off = (off + bytes + 255) & ~(size_t)255;
        return p;
    };
    int*    bar     = (int*)take(256);
    float*  h2      = (float*)take(2 * HDIM * sizeof(float));
    float*  cst     = (float*)take(HDIM * sizeof(float));
    __bf16* encEmb  = (__bf16*)take((size_t)TLEN * EMDIM * 2);
    __bf16* decEmb  = (__bf16*)take((size_t)TLEN * EMDIM * 2);
    __bf16* eWih    = (__bf16*)take((size_t)G4 * EMDIM * 2);
    __bf16* dWih    = (__bf16*)take((size_t)G4 * EMDIM * 2);
    __bf16* eWhh    = (__bf16*)take((size_t)G4 * HDIM * 2);
    __bf16* dWhh    = (__bf16*)take((size_t)G4 * HDIM * 2);
    __bf16* oW      = (__bf16*)take((size_t)OSIZE * HDIM * 2);
    float*  xgEnc   = (float*)take((size_t)TLEN * G4 * sizeof(float));
    float*  xgDec   = (float*)take((size_t)TLEN * G4 * sizeof(float));
    __bf16* hsB     = (__bf16*)take((size_t)TLEN * HDIM * 2);

    // 0) init barrier + state
    init_state_kernel<<<8, 256, 0, stream>>>(bar, h2, cst);

    // 1) weight conversions to bf16
    f2bf_kernel<<<2048, 256, 0, stream>>>(enc_Wih, eWih, (long)G4 * EMDIM);
    f2bf_kernel<<<2048, 256, 0, stream>>>(dec_Wih, dWih, (long)G4 * EMDIM);
    f2bf_kernel<<<2048, 256, 0, stream>>>(enc_Whh, eWhh, (long)G4 * HDIM);
    f2bf_kernel<<<2048, 256, 0, stream>>>(dec_Whh, dWhh, (long)G4 * HDIM);
    f2bf_kernel<<<2048, 256, 0, stream>>>(outW,    oW,   (long)OSIZE * HDIM);

    // 2) embedding gathers
    gather_enc_kernel<<<TLEN, 256, 0, stream>>>(emb_i, e1, encEmb);
    gather_dec_kernel<<<TLEN, 256, 0, stream>>>(emb_o, gt, decEmb);

    // 3) input-gate GEMMs: x_gates = embs @ Wih^T + (bih + bhh)
    wmma_gemm_bf16<<<dim3(G4 / 256, TLEN / 64), 256, 0, stream>>>(
        encEmb, eWih, enc_bih, enc_bhh, xgEnc, TLEN, G4, EMDIM);
    wmma_gemm_bf16<<<dim3(G4 / 256, TLEN / 64), 256, 0, stream>>>(
        decEmb, dWih, dec_bih, dec_bhh, xgDec, TLEN, G4, EMDIM);

    // 4) encoder recurrence (h0=c0=0); final cell state remains in cst
    lstm_persistent_kernel<<<LSTM_BLOCKS, 256, 0, stream>>>(
        xgEnc, eWhh, h2, cst, (__bf16*)nullptr, TLEN, bar);

    // 5) decoder init: h0 = c0 = enc_c
    dec_init_kernel<<<4, 256, 0, stream>>>(cst, h2);

    // 6) decoder recurrence with teacher forcing; emits hs in bf16
    lstm_persistent_kernel<<<LSTM_BLOCKS, 256, 0, stream>>>(
        xgDec, dWhh, h2, cst, hsB, TLEN, bar);

    // 7) logits = hs @ outW^T + outb   (dominant GEMM, fp32 out)
    wmma_gemm_bf16<<<dim3(OSIZE / 256, TLEN / 64), 256, 0, stream>>>(
        hsB, oW, outb, (const float*)nullptr, out, TLEN, OSIZE, HDIM);
}
